// GraphCON_GAT_5274219839933
// MI455X (gfx1250) — matrix operand order repaired and checked by reading, not compiled, not run on hardware
//
#include <hip/hip_runtime.h>

// ---------------------------------------------------------------------------
// GraphCON + sparse GAT on MI455X (gfx1250, wave32, WMMA).
// GEMMs: bf16 inputs, f32 accumulation via v_wmma_f32_16x16x32_bf16, with the
// A strip staged block-wide into LDS using async global->LDS copies
// (ASYNCcnt) and read back via ds_load_b128.
// Edge softmax-sums: global_atomic_add_f32.
// ---------------------------------------------------------------------------

#define NN      20000   // nodes
#define NE      256000  // edges
#define NF      128     // feature dim (NFEAT == NHID)
#define NHEADS  4
#define NSTEPS  3
#define NGRAPHS 64
#define ALPHA   0.2f

#define ROWS_PER_BLOCK 32          // 2 row tiles per wave
#define LDS_STRIDE     136         // 128 + 8 bf16 pad (16B) -> 4-bank row stagger

typedef __attribute__((ext_vector_type(16))) __bf16 v16bf;
typedef __attribute__((ext_vector_type(8)))  __bf16 v8bf;
typedef __attribute__((ext_vector_type(8)))  float  v8f;

// ---------------------------------------------------------------------------
// WMMA GEMM: C[M x 128] = A[M x 128](bf16, row-major) * B (bf16, pre-transposed
// as BT[n][k]) + optional bias.
// Block = 256 threads (8 waves) -> 32 rows x 128 cols. Each wave owns one
// 16-wide column tile and computes two 16x16 output tiles, reusing its B
// fragment across both. A strip (32x128 bf16 = 8KB) is async-copied to LDS
// once per block.
// ---------------------------------------------------------------------------
__global__ __launch_bounds__(256) void wmma_gemm_bf16(
    const __bf16* __restrict__ A,    // [M x 128]
    const __bf16* __restrict__ BT,   // [128 x 128], BT[n*128 + k] = B[k][n]
    const float*  __restrict__ bias, // [128] or nullptr
    float* __restrict__ C,           // [M x 128]
    int M)
{
    __shared__ __align__(16) __bf16 sA[ROWS_PER_BLOCK * LDS_STRIDE];

    const int tid  = threadIdx.x;
    const int lane = tid & 31;
    const int wave = tid >> 5;                  // column tile 0..7
    const int row0 = blockIdx.x * ROWS_PER_BLOCK;
    if (row0 >= M) return;                      // uniform per block

    // ---- stage A strip into LDS: 512 x 16B async transfers, 2 per thread ----
    {
        const char* gbase = (const char*)(A + (size_t)row0 * NF);
        const char* lbase = (const char*)sA;
#pragma unroll
        for (int i = 0; i < 2; ++i) {
            int t  = tid + i * 256;             // transfer id 0..511
            int r  = t >> 4;                    // source row 0..31
            int cb = (t & 15) * 16;             // byte offset within 256B row
            unsigned int       ldsoff = (unsigned int)(uintptr_t)(lbase + r * (LDS_STRIDE * 2) + cb);
            unsigned long long gaddr  = (unsigned long long)(uintptr_t)(gbase + r * 256 + cb);
            asm volatile("global_load_async_to_lds_b128 %0, %1, off"
                         :: "v"(ldsoff), "v"(gaddr) : "memory");
        }
        asm volatile("s_wait_asynccnt 0x0" ::: "memory");
        __syncthreads();
    }

    const int half = lane >> 4;                 // 0 or 1
    const int l15  = lane & 15;

    const __bf16* Bcol = BT + (size_t)(wave * 16 + l15) * NF;  // B: lane = col N
    const __bf16* a0r  = sA + (size_t)l15 * LDS_STRIDE;        // tile 0: rows 0..15
    const __bf16* a1r  = sA + (size_t)(16 + l15) * LDS_STRIDE; // tile 1: rows 16..31

    v8f acc0 = {}, acc1 = {};
#pragma unroll
    for (int k0 = 0; k0 < NF; k0 += 32) {
        // B 32x16 layout: lanes 0-15 hold K=[0..7],[16..23]; lanes 16-31 K=[8..15],[24..31].
        v8bf b_lo = *(const v8bf*)(Bcol + k0 + half * 8);
        v8bf b_hi = *(const v8bf*)(Bcol + k0 + 16 + half * 8);
        v16bf b = __builtin_shufflevector(b_lo, b_hi, 0,1,2,3,4,5,6,7,8,9,10,11,12,13,14,15);

        // A 16x32 layout (ISA 7.12.2): lanes 0-15 hold K=[0..7],[16..23];
        // lanes 16-31 hold K=[8..15],[24..31] for the same M.
        {
            v8bf alo = *(const v8bf*)(a0r + k0 + half * 8);
            v8bf ahi = *(const v8bf*)(a0r + k0 + 16 + half * 8);
            v16bf a = __builtin_shufflevector(alo, ahi, 0,1,2,3,4,5,6,7,8,9,10,11,12,13,14,15);
            acc0 = __builtin_amdgcn_wmma_f32_16x16x32_bf16(
                false, a, false, b, (short)0, acc0, false, false);
        }
        {
            v8bf alo = *(const v8bf*)(a1r + k0 + half * 8);
            v8bf ahi = *(const v8bf*)(a1r + k0 + 16 + half * 8);
            v16bf a = __builtin_shufflevector(alo, ahi, 0,1,2,3,4,5,6,7,8,9,10,11,12,13,14,15);
            acc1 = __builtin_amdgcn_wmma_f32_16x16x32_bf16(
                false, a, false, b, (short)0, acc1, false, false);
        }
    }

    const int   c  = wave * 16 + l15;
    const float bv = bias ? bias[c] : 0.0f;
    // C/D layout: VGPR v -> M = v + 8*half, N = lane&15 (+ column tile base).
#pragma unroll
    for (int v = 0; v < 8; ++v) {
        const int r = row0 + v + 8 * half;
        C[(size_t)r * NF + c]        = acc0[v] + bv;
        C[(size_t)(r + 16) * NF + c] = acc1[v] + bv;
    }
}

// ---------------------------------------------------------------------------
// Prep kernels
// ---------------------------------------------------------------------------
__global__ void cvt_f32_bf16(const float* __restrict__ in, __bf16* __restrict__ out, int n)
{
    int i = blockIdx.x * blockDim.x + threadIdx.x;
    if (i < n) out[i] = (__bf16)in[i];
}

// out[m][n][k] = in[m][k][n]  (transpose each 128x128 matrix, f32 -> bf16)
__global__ void transpose_cvt_bf16(const float* __restrict__ in, __bf16* __restrict__ out, int nmat)
{
    int idx = blockIdx.x * blockDim.x + threadIdx.x;
    int total = nmat * NF * NF;
    if (idx >= total) return;
    int m = idx / (NF * NF);
    int r = idx - m * NF * NF;
    int n = r / NF, k = r - n * NF;
    out[idx] = (__bf16)in[(size_t)m * NF * NF + (size_t)k * NF + n];
}

// wa[h][0][i] = sum_j W_att[h][i][j] * a_att[h][j]        (first half of a)
// wa[h][1][i] = sum_j W_att[h][i][j] * a_att[h][128 + j]  (second half)
__global__ void prep_wa(const float* __restrict__ W_att, const float* __restrict__ a_att,
                        float* __restrict__ wa)
{
    int gid  = blockIdx.x * (blockDim.x >> 5) + (threadIdx.x >> 5);  // 0..511
    if (gid >= NHEADS * NF) return;
    int head = gid >> 7, i = gid & (NF - 1);
    int lane = threadIdx.x & 31;
    const float* Wrow = W_att + ((size_t)head * NF + i) * NF;
    const float* a1 = a_att + head * 2 * NF;
    const float* a2 = a1 + NF;
    float d1 = 0.f, d2 = 0.f;
#pragma unroll
    for (int j = lane; j < NF; j += 32) {
        float w = Wrow[j];
        d1 = fmaf(w, a1[j], d1);
        d2 = fmaf(w, a2[j], d2);
    }
#pragma unroll
    for (int off = 16; off; off >>= 1) {
        d1 += __shfl_xor(d1, off, 32);
        d2 += __shfl_xor(d2, off, 32);
    }
    if (lane == 0) {
        wa[(head * 2 + 0) * NF + i] = d1;
        wa[(head * 2 + 1) * NF + i] = d2;
    }
}

// y = h; hbf = bf16(h)
__global__ void init_y_hbf(const float* __restrict__ h, float* __restrict__ y,
                           __bf16* __restrict__ hbf, int n)
{
    int i = blockIdx.x * blockDim.x + threadIdx.x;
    if (i < n) { float v = h[i]; y[i] = v; hbf[i] = (__bf16)v; }
}

// ---------------------------------------------------------------------------
// Per-node logit projections: s1 = h . wa1 ; s2 = h . wa2  (one wave / node)
// ---------------------------------------------------------------------------
__global__ void node_dots(const float* __restrict__ h, const float* __restrict__ wa1,
                          const float* __restrict__ wa2, float* __restrict__ s1,
                          float* __restrict__ s2, int n_nodes)
{
    int node = blockIdx.x * (blockDim.x >> 5) + (threadIdx.x >> 5);
    if (node >= n_nodes) return;
    int lane = threadIdx.x & 31;
    const float4 x = ((const float4*)(h + (size_t)node * NF))[lane];
    const float4 a = ((const float4*)wa1)[lane];
    const float4 b = ((const float4*)wa2)[lane];
    float d1 = x.x * a.x + x.y * a.y + x.z * a.z + x.w * a.w;
    float d2 = x.x * b.x + x.y * b.y + x.z * b.z + x.w * b.w;
#pragma unroll
    for (int off = 16; off; off >>= 1) {
        d1 += __shfl_xor(d1, off, 32);
        d2 += __shfl_xor(d2, off, 32);
    }
    if (lane == 0) { s1[node] = d1; s2[node] = d2; }
}

// ---------------------------------------------------------------------------
// Edge scatter: w = exp(-leaky_relu(s1[src] + s2[dst]));
// rowsum[src] += w; hp[src,:] += w * hh[dst,:]   (one wave / edge, float4/lane)
// ---------------------------------------------------------------------------
__global__ void edge_scatter(const int* __restrict__ src, const int* __restrict__ dst,
                             const float* __restrict__ s1, const float* __restrict__ s2,
                             const float* __restrict__ hh, float* __restrict__ rowsum,
                             float* __restrict__ hp, int n_edges)
{
    int e = blockIdx.x * (blockDim.x >> 5) + (threadIdx.x >> 5);
    if (e >= n_edges) return;
    int lane = threadIdx.x & 31;
    int s = src[e], d = dst[e];
    float logit = s1[s] + s2[d];
    float lr = logit > 0.0f ? logit : ALPHA * logit;
    float w = __expf(-lr);
    if (lane == 0) atomicAdd(&rowsum[s], w);
    float4 hv = ((const float4*)(hh + (size_t)d * NF))[lane];
    float* hpb = hp + (size_t)s * NF + lane * 4;
    atomicAdd(hpb + 0, w * hv.x);
    atomicAdd(hpb + 1, w * hv.y);
    atomicAdd(hpb + 2, w * hv.z);
    atomicAdd(hpb + 3, w * hv.w);
}

// ---------------------------------------------------------------------------
// att (+)= elu(hp / (rowsum + 1e-16) + h)
// ---------------------------------------------------------------------------
__global__ void att_accum(const float* __restrict__ hp, const float* __restrict__ rowsum,
                          const float* __restrict__ h, float* __restrict__ att,
                          int first, int n)
{
    int idx = blockIdx.x * blockDim.x + threadIdx.x;
    if (idx >= n) return;
    int node = idx >> 7;
    float v = hp[idx] / (rowsum[node] + 1e-16f) + h[idx];
    float el = v > 0.0f ? v : (__expf(v) - 1.0f);
    att[idx] = first ? el : att[idx] + el;
}

// ---------------------------------------------------------------------------
// GraphCON oscillator update (dt = diff = freq = 1):
// h' = h + (att/4 - h - y); y' = y + h'
// ---------------------------------------------------------------------------
__global__ void graphcon_update(float* __restrict__ h, float* __restrict__ y,
                                const float* __restrict__ att, __bf16* __restrict__ hbf, int n)
{
    int idx = blockIdx.x * blockDim.x + threadIdx.x;
    if (idx >= n) return;
    float a  = att[idx] * 0.25f;
    float hv = h[idx], yv = y[idx];
    float hn = hv + (a - hv - yv);
    float yn = yv + hn;
    h[idx] = hn; y[idx] = yn; hbf[idx] = (__bf16)hn;
}

// ---------------------------------------------------------------------------
// out = y @ W_out + b_out; pooled[batch[node]] += out   (one wave / node)
// ---------------------------------------------------------------------------
__global__ void out_pool(const float* __restrict__ y, const float* __restrict__ W_out,
                         const float* __restrict__ b_out, const int* __restrict__ batch,
                         float* __restrict__ pooled, int n_nodes)
{
    int node = blockIdx.x * (blockDim.x >> 5) + (threadIdx.x >> 5);
    if (node >= n_nodes) return;
    int lane = threadIdx.x & 31;
    float4 a = ((const float4*)(y + (size_t)node * NF))[lane];
    float4 b = ((const float4*)W_out)[lane];
    float d = a.x * b.x + a.y * b.y + a.z * b.z + a.w * b.w;
#pragma unroll
    for (int off = 16; off; off >>= 1) d += __shfl_xor(d, off, 32);
    if (lane == 0) atomicAdd(&pooled[batch[node]], d + b_out[0]);
}

// ---------------------------------------------------------------------------
// Host-side orchestration
// ---------------------------------------------------------------------------
static inline char* ws_carve(char*& p, size_t bytes)
{
    char* r = p;
    p += (bytes + 255) & ~size_t(255);
    return r;
}

extern "C" void kernel_launch(void* const* d_in, const int* in_sizes, int n_in,
                              void* d_out, int out_size, void* d_ws, size_t ws_size,
                              hipStream_t stream)
{
    (void)in_sizes; (void)n_in; (void)out_size; (void)ws_size;

    const float* x      = (const float*)d_in[0];
    const int*   edge   = (const int*)  d_in[1];
    const int*   src    = edge;
    const int*   dst    = edge + NE;
    const int*   batch  = (const int*)  d_in[2];
    const float* W_emb  = (const float*)d_in[3];
    const float* b_emb  = (const float*)d_in[4];
    const float* W_att  = (const float*)d_in[5];
    const float* a_att  = (const float*)d_in[6];
    const float* W_out  = (const float*)d_in[7];
    const float* b_out  = (const float*)d_in[8];
    float* pooled = (float*)d_out;

    // Workspace carve-out (~62 MB, all resident in L2)
    char* p = (char*)d_ws;
    float*  h      = (float*) ws_carve(p, (size_t)NN * NF * 4);
    float*  y      = (float*) ws_carve(p, (size_t)NN * NF * 4);
    float*  att    = (float*) ws_carve(p, (size_t)NN * NF * 4);
    float*  hh     = (float*) ws_carve(p, (size_t)NN * NF * 4);
    float*  hp     = (float*) ws_carve(p, (size_t)NN * NF * 4);
    float*  rowsum = (float*) ws_carve(p, (size_t)NN * 4);
    float*  s1     = (float*) ws_carve(p, (size_t)NN * 4);
    float*  s2     = (float*) ws_carve(p, (size_t)NN * 4);
    float*  wa     = (float*) ws_carve(p, (size_t)NHEADS * 2 * NF * 4);
    __bf16* xbf    = (__bf16*)ws_carve(p, (size_t)NN * NF * 2);
    __bf16* hbf    = (__bf16*)ws_carve(p, (size_t)NN * NF * 2);
    __bf16* WembT  = (__bf16*)ws_carve(p, (size_t)NF * NF * 2);
    __bf16* WattT  = (__bf16*)ws_carve(p, (size_t)NHEADS * NF * NF * 2);

    const int ELEMS     = NN * NF;                  // 2,560,000
    const int EL_BLOCKS = (ELEMS + 255) / 256;      // 10,000
    const int GEMM_GRID = NN / ROWS_PER_BLOCK;      // 625 (32-row strips)
    const int NODE_WAVE = (NN + 7) / 8;             // 2,500 (8 waves/block)
    const int EDGE_GRID = (NE + 7) / 8;             // 32,000

    // --- prep: bf16 conversions + transposed weights + W@a vectors ---
    cvt_f32_bf16<<<EL_BLOCKS, 256, 0, stream>>>(x, xbf, ELEMS);
    transpose_cvt_bf16<<<(NF * NF + 255) / 256, 256, 0, stream>>>(W_emb, WembT, 1);
    transpose_cvt_bf16<<<(NHEADS * NF * NF + 255) / 256, 256, 0, stream>>>(W_att, WattT, NHEADS);
    prep_wa<<<64, 256, 0, stream>>>(W_att, a_att, wa);

    // --- embedding: h = x @ W_emb + b_emb ; y = h ---
    wmma_gemm_bf16<<<GEMM_GRID, 256, 0, stream>>>(xbf, WembT, b_emb, h, NN);
    init_y_hbf<<<EL_BLOCKS, 256, 0, stream>>>(h, y, hbf, ELEMS);

    // --- GraphCON steps ---
    for (int step = 0; step < NSTEPS; ++step) {
        for (int head = 0; head < NHEADS; ++head) {
            wmma_gemm_bf16<<<GEMM_GRID, 256, 0, stream>>>(
                hbf, WattT + (size_t)head * NF * NF, nullptr, hh, NN);
            node_dots<<<NODE_WAVE, 256, 0, stream>>>(
                h, wa + (head * 2 + 0) * NF, wa + (head * 2 + 1) * NF, s1, s2, NN);
            hipMemsetAsync(rowsum, 0, (size_t)NN * 4, stream);
            hipMemsetAsync(hp, 0, (size_t)NN * NF * 4, stream);
            edge_scatter<<<EDGE_GRID, 256, 0, stream>>>(
                src, dst, s1, s2, hh, rowsum, hp, NE);
            att_accum<<<EL_BLOCKS, 256, 0, stream>>>(
                hp, rowsum, h, att, head == 0 ? 1 : 0, ELEMS);
        }
        graphcon_update<<<EL_BLOCKS, 256, 0, stream>>>(h, y, att, hbf, ELEMS);
    }

    // --- readout + global_add_pool ---
    hipMemsetAsync(pooled, 0, (size_t)NGRAPHS * 4, stream);
    out_pool<<<NODE_WAVE, 256, 0, stream>>>(y, W_out, b_out, batch, pooled, NN);
}
